// Kernel_21260088115409
// MI455X (gfx1250) — matrix operand order repaired and checked by reading, not compiled, not run on hardware
//
#include <hip/hip_runtime.h>
#include <hip/hip_bf16.h>

// Problem constants (from reference): x:[N,D], y:[M,D], out:[N,M], fp32.
#define NN 4096
#define MMM 4096
#define DD 1024
#define BK 32      // K chunk = bf16 WMMA K depth
#define BT 128     // workgroup output tile: 128x128
#define LSTR 36    // LDS row stride in bf16 elements (72B: 8B-aligned, bank-friendly)

typedef __attribute__((ext_vector_type(16))) __bf16 v16bf;
typedef __attribute__((ext_vector_type(8)))  float  v8f;
typedef __attribute__((ext_vector_type(4)))  __bf16 bf16x4;

// ---------------- row squared-norms: one wave per row ----------------
__global__ void __launch_bounds__(256) rbf_row_norms(const float* __restrict__ x,
                                                     const float* __restrict__ y,
                                                     float* __restrict__ ws) {
    const int wid  = (blockIdx.x * blockDim.x + threadIdx.x) >> 5;  // 0..8191
    const int lane = threadIdx.x & 31;
    const float4* src = (wid < NN)
        ? (const float4*)(x + (size_t)wid * DD)
        : (const float4*)(y + (size_t)(wid - NN) * DD);
    float s = 0.f;
#pragma unroll
    for (int i = 0; i < DD / 128; ++i) {            // 8 x float4 per lane
        float4 v = src[i * 32 + lane];
        s += v.x * v.x + v.y * v.y + v.z * v.z + v.w * v.w;
    }
#pragma unroll
    for (int off = 16; off > 0; off >>= 1) s += __shfl_down(s, off, 32);
    if (lane == 0) ws[wid] = s;
}

// ------------- helpers -------------
__device__ __forceinline__ void cvt4(float4 f, bf16x4& h, bf16x4& l) {
    float a0 = f.x, a1 = f.y, a2 = f.z, a3 = f.w;
    __bf16 h0 = (__bf16)a0, h1 = (__bf16)a1, h2 = (__bf16)a2, h3 = (__bf16)a3;
    h[0] = h0; h[1] = h1; h[2] = h2; h[3] = h3;
    l[0] = (__bf16)(a0 - (float)h0);
    l[1] = (__bf16)(a1 - (float)h1);
    l[2] = (__bf16)(a2 - (float)h2);
    l[3] = (__bf16)(a3 - (float)h3);
}

// A fragment (16-bit A 16x32): lane<16 holds K {0..7,16..23}; lane>=16 holds {8..15,24..31}.
// base already includes (row*LSTR + k0a) with k0a = (lane>>4)*8.
__device__ __forceinline__ v16bf lds_frag_a(const __bf16* base) {
    union { unsigned long long u[4]; v16bf v; } f;
    const unsigned long long* q = (const unsigned long long*)base;
    f.u[0] = q[0];      // K0..K0+3
    f.u[1] = q[1];      // K0+4..K0+7
    f.u[2] = q[4];      // K0+16..K0+19
    f.u[3] = q[5];      // K0+20..K0+23
    return f.v;
}

// B fragment (16-bit B 32x16): lane<16 -> col=lane, K {0..7,16..23}; lane>=16 -> K {8..15,24..31}.
// base already includes (col*LSTR + k0b) with k0b = (lane>>4)*8.
__device__ __forceinline__ v16bf lds_frag_b(const __bf16* base) {
    union { unsigned long long u[4]; v16bf v; } f;
    const unsigned long long* q = (const unsigned long long*)base;
    f.u[0] = q[0]; f.u[1] = q[1]; f.u[2] = q[4]; f.u[3] = q[5];
    return f.v;
}

// ---------------- fused RBF GEMM ----------------
__global__ void __launch_bounds__(256) rbf_wmma(const float* __restrict__ x,
                                                const float* __restrict__ y,
                                                const float* __restrict__ ws,
                                                const float* __restrict__ log_sigma,
                                                float* __restrict__ out) {
    __shared__ __align__(16) __bf16 Ah[BT * LSTR];
    __shared__ __align__(16) __bf16 Al[BT * LSTR];
    __shared__ __align__(16) __bf16 Bh[BT * LSTR];
    __shared__ __align__(16) __bf16 Bl[BT * LSTR];

    const int t    = threadIdx.x;
    const int lane = t & 31;
    const int w    = t >> 5;        // 8 waves
    const int wrow = w & 3;         // 4 row slabs of 32
    const int wcol = w >> 2;        // 2 col slabs of 64
    const int rowBase = blockIdx.y * BT;
    const int colBase = blockIdx.x * BT;

    // cooperative loader mapping: 256 threads cover 128 rows x 32 K
    const int lr = t >> 1;          // 0..127
    const int lc = (t & 1) * 16;    // 0 or 16

    const float* xrow = x + (size_t)(rowBase + lr) * DD + lc;
    const float* yrow = y + (size_t)(colBase + lr) * DD + lc;
    __bf16* ahp = Ah + lr * LSTR + lc;
    __bf16* alp = Al + lr * LSTR + lc;
    __bf16* bhp = Bh + lr * LSTR + lc;
    __bf16* blp = Bl + lr * LSTR + lc;

    // per-lane fragment offsets (see loaders above)
    const int fr   = lane & 15;
    const int aoff = fr * LSTR + ((lane >> 4) << 3);
    const int boff = fr * LSTR + ((lane >> 4) << 3);

    v8f acc[2][4];
#pragma unroll
    for (int i = 0; i < 2; ++i)
#pragma unroll
        for (int j = 0; j < 4; ++j) acc[i][j] = (v8f)0.0f;

    for (int k0 = 0; k0 < DD; k0 += BK) {
#pragma unroll
        for (int j = 0; j < 4; ++j) {
            float4 vx = *(const float4*)(xrow + k0 + 4 * j);
            float4 vy = *(const float4*)(yrow + k0 + 4 * j);
            bf16x4 hx, lx, hy, ly;
            cvt4(vx, hx, lx);
            cvt4(vy, hy, ly);
            *(bf16x4*)(ahp + 4 * j) = hx;
            *(bf16x4*)(alp + 4 * j) = lx;
            *(bf16x4*)(bhp + 4 * j) = hy;
            *(bf16x4*)(blp + 4 * j) = ly;
        }
        __syncthreads();

        v16bf ah[2], al[2], bh[4], bl[4];
#pragma unroll
        for (int i = 0; i < 2; ++i) {
            const int o = (wrow * 32 + i * 16) * LSTR + aoff;
            ah[i] = lds_frag_a(Ah + o);
            al[i] = lds_frag_a(Al + o);
        }
#pragma unroll
        for (int j = 0; j < 4; ++j) {
            const int o = (wcol * 64 + j * 16) * LSTR + boff;
            bh[j] = lds_frag_b(Bh + o);
            bl[j] = lds_frag_b(Bl + o);
        }

#pragma unroll
        for (int i = 0; i < 2; ++i)
#pragma unroll
            for (int j = 0; j < 4; ++j) {
                // split-bf16 fp32 emulation: xh*yh + xh*yl + xl*yh
                acc[i][j] = __builtin_amdgcn_wmma_f32_16x16x32_bf16(
                    false, ah[i], false, bh[j], (short)0, acc[i][j], false, false);
                acc[i][j] = __builtin_amdgcn_wmma_f32_16x16x32_bf16(
                    false, ah[i], false, bl[j], (short)0, acc[i][j], false, false);
                acc[i][j] = __builtin_amdgcn_wmma_f32_16x16x32_bf16(
                    false, al[i], false, bh[j], (short)0, acc[i][j], false, false);
            }
        __syncthreads();
    }

    // ---------------- epilogue: FK = xx + yy - 2*dot ; out = exp(-FK * 1/(2 sigma^2))
    const float ls = *log_sigma;
    const float sc = 0.5f * __expf(-2.0f * ls);   // 1/(2*sigma^2)
    const float* xx = ws;
    const float* yy = ws + NN;
    const int lh = lane >> 4;

#pragma unroll
    for (int i = 0; i < 2; ++i) {
        const int rb = rowBase + wrow * 32 + i * 16 + lh * 8;
#pragma unroll
        for (int j = 0; j < 4; ++j) {
            const int cb = colBase + wcol * 64 + j * 16 + fr;
            const float yn = yy[cb];
            v8f a = acc[i][j];
#pragma unroll
            for (int v = 0; v < 8; ++v) {
                const int row = rb + v;
                const float fk = xx[row] + yn - 2.0f * a[v];
                out[(size_t)row * MMM + cb] = __expf(-fk * sc);
            }
        }
    }
}

extern "C" void kernel_launch(void* const* d_in, const int* in_sizes, int n_in,
                              void* d_out, int out_size, void* d_ws, size_t ws_size,
                              hipStream_t stream) {
    const float* x  = (const float*)d_in[0];
    const float* y  = (const float*)d_in[1];
    const float* ls = (const float*)d_in[2];   // single-element log_sigma
    float* out = (float*)d_out;
    float* ws  = (float*)d_ws;                 // needs (N+M)*4 = 32 KB

    // 1) row norms: 8192 rows, one wave per row, 8 waves per block
    rbf_row_norms<<<(NN + MMM) / 8, 256, 0, stream>>>(x, y, ws);

    // 2) fused split-bf16 WMMA GEMM + RBF epilogue
    dim3 grid(MMM / BT, NN / BT);
    rbf_wmma<<<grid, 256, 0, stream>>>(x, y, ws, ls, out);
}
